// SentenceRewardModel_38955353374927
// MI455X (gfx1250) — matrix operand order repaired and checked by reading, not compiled, not run on hardware
//
#include <hip/hip_runtime.h>
#include <hip/hip_bf16.h>

// Problem constants (match reference setup)
#define BB 8
#define SS 2048
#define HH 4096
#define DD 128
#define NSENT 128

typedef __attribute__((ext_vector_type(16))) _Float16 v16h;
typedef __attribute__((ext_vector_type(8)))  _Float16 v8h;
typedef __attribute__((ext_vector_type(8)))  float    v8f;

// ln(10000)/64
#define LOG10000_OVER_64 0.1439115683121281f

static __device__ __forceinline__ v16h cat16(v8h lo, v8h hi) {
    union { v16h v; v8h h[2]; } u;
    u.h[0] = lo; u.h[1] = hi;
    return u.v;
}

// ---------------------------------------------------------------------------
// Kernel 1: positions of the NS ones per row, ascending (stable serial scan)
// ---------------------------------------------------------------------------
__global__ void build_idx_kernel(const int* __restrict__ mask, int* __restrict__ idx) {
    int b = blockIdx.x;
    if (threadIdx.x != 0) return;
    int c = 0;
    for (int s = 0; s < SS; ++s) {
        if (mask[b * SS + s] != 0 && c < NSENT) idx[b * NSENT + c++] = s;
    }
}

// ---------------------------------------------------------------------------
// Kernel 2: Wqk_f16[256][4096] = concat(Wq, Wk) converted to f16
// ---------------------------------------------------------------------------
__global__ void convert_w_kernel(const float* __restrict__ Wq, const float* __restrict__ Wk,
                                 _Float16* __restrict__ Wqk) {
    size_t total = (size_t)2 * DD * HH;
    for (size_t i = (size_t)blockIdx.x * blockDim.x + threadIdx.x; i < total;
         i += (size_t)gridDim.x * blockDim.x) {
        size_t r = i / HH, h = i % HH;
        float x = (r < DD) ? Wq[r * HH + h] : Wk[(r - DD) * HH + h];
        Wqk[i] = (_Float16)x;
    }
}

// ---------------------------------------------------------------------------
// Kernel 3: gather hidden at idx, convert to f16, fused v = Wr.x + br
// grid (NSENT, BB) x 256 threads; each thread handles 16 contiguous floats.
// Prefetch the NEXT gathered row (random 16KB lines) while converting this one.
// ---------------------------------------------------------------------------
__global__ __launch_bounds__(256)
void gather_convert_kernel(const float* __restrict__ hs, const int* __restrict__ idx,
                           const float* __restrict__ Wr, const float* __restrict__ br,
                           _Float16* __restrict__ Xh, float* __restrict__ v) {
    const int n = blockIdx.x, b = blockIdx.y, t = threadIdx.x;
    const int tok = idx[b * NSENT + n];
    const float* src = hs + ((size_t)b * SS + tok) * HH;
    _Float16* dst = Xh + ((size_t)(b * NSENT + n)) * HH;

    // prefetch this thread's slice of the next sentence's row (global_prefetch_b8)
    if (n + 1 < NSENT) {
        const int tok_next = idx[b * NSENT + n + 1];
        __builtin_prefetch(hs + ((size_t)b * SS + tok_next) * HH + t * 16, 0, 1);
    }

    float acc = 0.f;
    #pragma unroll
    for (int i = t * 16; i < t * 16 + 16; i += 4) {
        float4 x = *(const float4*)(src + i);
        float4 w = *(const float4*)(Wr + i);
        dst[i]     = (_Float16)x.x;
        dst[i + 1] = (_Float16)x.y;
        dst[i + 2] = (_Float16)x.z;
        dst[i + 3] = (_Float16)x.w;
        acc += x.x * w.x + x.y * w.y + x.z * w.z + x.w * w.w;
    }
    __shared__ float red[256];
    red[t] = acc;
    __syncthreads();
    for (int off = 128; off > 0; off >>= 1) {
        if (t < off) red[t] += red[t + off];
        __syncthreads();
    }
    if (t == 0) v[b * NSENT + n] = red[0] + br[0];
}

// ---------------------------------------------------------------------------
// Kernel 4: WMMA GEMM  QK[b][128][256] = Xh[b] (128x4096, f16) * Wqk^T + bias
// One wave computes a 16x64 strip (4 accumulators). Explicit double-buffered
// software pipeline: buffer X holds chunk k, buffer Y chunk k+32; each WMMA
// group only waits on loads issued one full iteration earlier. grid (4, 8, B).
// A (16x32 f16) layout: lane l holds row m0+(l&15); 16 contiguous K at +(l>>4)*16
// B (32x16 f16) layout: lane l holds col n+(l&15); 16 contiguous K at +(l>>4)*16
// ---------------------------------------------------------------------------
__global__ __launch_bounds__(32)
void gemm_qk_kernel(const _Float16* __restrict__ Xh, const _Float16* __restrict__ Wqk,
                    const float* __restrict__ bq, const float* __restrict__ bk,
                    float* __restrict__ QK) {
    const int lane  = threadIdx.x;
    const int n0    = blockIdx.x * 64;   // 4 col tiles of 16
    const int m0    = blockIdx.y * 16;
    const int b     = blockIdx.z;
    const int lhalf = lane >> 4;   // 0 or 1
    const int lmod  = lane & 15;

    const _Float16* arow  = Xh  + ((size_t)(b * NSENT + m0 + lmod)) * HH + lhalf * 16;
    const _Float16* brow0 = Wqk + ((size_t)(n0 + lmod)) * HH + lhalf * 16;
    const _Float16* brow1 = brow0 + (size_t)16 * HH;
    const _Float16* brow2 = brow0 + (size_t)32 * HH;
    const _Float16* brow3 = brow0 + (size_t)48 * HH;

    // double-buffered operand sets
    v8h xal, xah, xb0l, xb0h, xb1l, xb1h, xb2l, xb2h, xb3l, xb3h;
    v8h yal, yah, yb0l, yb0h, yb1l, yb1h, yb2l, yb2h, yb3l, yb3h;

#define LOAD_X(K)                                                             \
    do {                                                                      \
        xal  = *(const v8h*)(arow  + (K));                                    \
        xah  = *(const v8h*)(arow  + (K) + 8);                                \
        xb0l = *(const v8h*)(brow0 + (K));                                    \
        xb0h = *(const v8h*)(brow0 + (K) + 8);                                \
        xb1l = *(const v8h*)(brow1 + (K));                                    \
        xb1h = *(const v8h*)(brow1 + (K) + 8);                                \
        xb2l = *(const v8h*)(brow2 + (K));                                    \
        xb2h = *(const v8h*)(brow2 + (K) + 8);                                \
        xb3l = *(const v8h*)(brow3 + (K));                                    \
        xb3h = *(const v8h*)(brow3 + (K) + 8);                                \
    } while (0)

#define LOAD_Y(K)                                                             \
    do {                                                                      \
        yal  = *(const v8h*)(arow  + (K));                                    \
        yah  = *(const v8h*)(arow  + (K) + 8);                                \
        yb0l = *(const v8h*)(brow0 + (K));                                    \
        yb0h = *(const v8h*)(brow0 + (K) + 8);                                \
        yb1l = *(const v8h*)(brow1 + (K));                                    \
        yb1h = *(const v8h*)(brow1 + (K) + 8);                                \
        yb2l = *(const v8h*)(brow2 + (K));                                    \
        yb2h = *(const v8h*)(brow2 + (K) + 8);                                \
        yb3l = *(const v8h*)(brow3 + (K));                                    \
        yb3h = *(const v8h*)(brow3 + (K) + 8);                                \
    } while (0)

#define WMMA_X()                                                                                  \
    do {                                                                                          \
        v16h a = cat16(xal, xah);                                                                 \
        c0 = __builtin_amdgcn_wmma_f32_16x16x32_f16(false, a, false, cat16(xb0l, xb0h), (short)0, \
                                                    c0, false, false);                            \
        c1 = __builtin_amdgcn_wmma_f32_16x16x32_f16(false, a, false, cat16(xb1l, xb1h), (short)0, \
                                                    c1, false, false);                            \
        c2 = __builtin_amdgcn_wmma_f32_16x16x32_f16(false, a, false, cat16(xb2l, xb2h), (short)0, \
                                                    c2, false, false);                            \
        c3 = __builtin_amdgcn_wmma_f32_16x16x32_f16(false, a, false, cat16(xb3l, xb3h), (short)0, \
                                                    c3, false, false);                            \
    } while (0)

#define WMMA_Y()                                                                                  \
    do {                                                                                          \
        v16h a = cat16(yal, yah);                                                                 \
        c0 = __builtin_amdgcn_wmma_f32_16x16x32_f16(false, a, false, cat16(yb0l, yb0h), (short)0, \
                                                    c0, false, false);                            \
        c1 = __builtin_amdgcn_wmma_f32_16x16x32_f16(false, a, false, cat16(yb1l, yb1h), (short)0, \
                                                    c1, false, false);                            \
        c2 = __builtin_amdgcn_wmma_f32_16x16x32_f16(false, a, false, cat16(yb2l, yb2h), (short)0, \
                                                    c2, false, false);                            \
        c3 = __builtin_amdgcn_wmma_f32_16x16x32_f16(false, a, false, cat16(yb3l, yb3h), (short)0, \
                                                    c3, false, false);                            \
    } while (0)

    v8f c0 = {}, c1 = {}, c2 = {}, c3 = {};

    LOAD_X(0);
    LOAD_Y(32);
    #pragma unroll 1
    for (int k0 = 0; k0 < HH - 64; k0 += 64) {
        WMMA_X();
        LOAD_X(k0 + 64);
        WMMA_Y();
        LOAD_Y(k0 + 96);
    }
    WMMA_X();   // chunk HH-64
    WMMA_Y();   // chunk HH-32

#undef LOAD_X
#undef LOAD_Y
#undef WMMA_X
#undef WMMA_Y

    // C/D layout: VGPR r -> row m0 + r + 8*lhalf, col = ntile + lmod
    float* outbase = QK + ((size_t)(b * NSENT + m0 + lhalf * 8)) * 256;
    #pragma unroll
    for (int j = 0; j < 4; ++j) {
        const int col = n0 + j * 16 + lmod;
        const float bias = (col < DD) ? bq[col] : bk[col - DD];
        const v8f c = (j == 0) ? c0 : (j == 1) ? c1 : (j == 2) ? c2 : c3;
        float* out = outbase + col;
        #pragma unroll
        for (int r = 0; r < 8; ++r) out[(size_t)r * 256] = c[r] + bias;
    }
}

// ---------------------------------------------------------------------------
// Kernel 5: per-batch RoPE + softmax + reward epilogue. <<<B, 128>>>
// QK cols [0,128) = q, [128,256) = k.
// ---------------------------------------------------------------------------
__global__ __launch_bounds__(128)
void finalize_kernel(const float* __restrict__ QK, const float* __restrict__ v,
                     float* __restrict__ out_sent, float* __restrict__ out_seq) {
    const int b = blockIdx.x;
    const int t = threadIdx.x;
    __shared__ float invf_s[DD / 2];
    __shared__ float qrot[DD];
    __shared__ float red[128];

    // inverse frequencies, computed once per block
    if (t < DD / 2) invf_s[t] = expf(-(float)t * LOG10000_OVER_64);
    __syncthreads();

    // RoPE-rotate q of the LAST sentence (position NSENT-1)
    if (t < DD / 2) {
        float ang = (float)(NSENT - 1) * invf_s[t];
        float s, c;
        sincosf(ang, &s, &c);
        const float* qrow = QK + ((size_t)(b * NSENT + NSENT - 1)) * 256;
        float q0 = qrow[2 * t], q1 = qrow[2 * t + 1];
        qrot[2 * t]     = q0 * c - q1 * s;
        qrot[2 * t + 1] = q0 * s + q1 * c;
    }
    __syncthreads();

    // thread t: RoPE-rotate k of sentence t (position t), dot with qrot
    const float* krow = QK + ((size_t)(b * NSENT + t)) * 256 + DD;
    float logit = 0.f;
    #pragma unroll 4
    for (int j = 0; j < DD / 2; ++j) {
        float ang = (float)t * invf_s[j];
        float s, c;
        sincosf(ang, &s, &c);
        float k0 = krow[2 * j], k1 = krow[2 * j + 1];
        float kr0 = k0 * c - k1 * s;
        float kr1 = k0 * s + k1 * c;
        logit += kr0 * qrot[2 * j] + kr1 * qrot[2 * j + 1];
    }
    logit *= rsqrtf((float)DD);

    // softmax over 128 logits
    red[t] = logit;
    __syncthreads();
    for (int off = 64; off > 0; off >>= 1) {
        if (t < off) red[t] = fmaxf(red[t], red[t + off]);
        __syncthreads();
    }
    float mx = red[0];
    __syncthreads();
    float e = expf(logit - mx);
    red[t] = e;
    __syncthreads();
    for (int off = 64; off > 0; off >>= 1) {
        if (t < off) red[t] += red[t + off];
        __syncthreads();
    }
    float denom = red[0];
    __syncthreads();
    float attn = e / denom;

    // sr = diff of v with first element kept
    float vv    = v[b * NSENT + t];
    float vprev = (t > 0) ? v[b * NSENT + t - 1] : 0.f;
    float srw   = (vv - vprev) * attn;
    out_sent[b * NSENT + t] = srw;

    red[t] = srw;
    __syncthreads();
    for (int off = 64; off > 0; off >>= 1) {
        if (t < off) red[t] += red[t + off];
        __syncthreads();
    }
    if (t == 0) out_seq[b] = red[0];
}

// ---------------------------------------------------------------------------
extern "C" void kernel_launch(void* const* d_in, const int* in_sizes, int n_in,
                              void* d_out, int out_size, void* d_ws, size_t ws_size,
                              hipStream_t stream) {
    const float* hs   = (const float*)d_in[0];
    const int*   mask = (const int*)d_in[1];
    // d_in[2] = num_sentences (always 128 here)
    const float* Wq = (const float*)d_in[3];
    const float* bq = (const float*)d_in[4];
    const float* Wk = (const float*)d_in[5];
    const float* bk = (const float*)d_in[6];
    const float* Wr = (const float*)d_in[7];
    const float* br = (const float*)d_in[8];

    char* ws = (char*)d_ws;
    // workspace layout (16B-aligned offsets)
    int*      idx  = (int*)(ws);                               //   4 KB
    float*    vbuf = (float*)(ws + 4096);                      //   4 KB
    _Float16* Wqk  = (_Float16*)(ws + 8192);                   //   2 MB
    _Float16* Xh   = (_Float16*)(ws + 8192 + (size_t)2 * DD * HH * 2);            // 8 MB
    float*    QK   = (float*)(ws + 8192 + (size_t)2 * DD * HH * 2
                                   + (size_t)BB * NSENT * HH * 2);                // 1 MB

    float* out_sent = (float*)d_out;            // [B, NS]
    float* out_seq  = out_sent + BB * NSENT;    // [B]

    build_idx_kernel<<<BB, 32, 0, stream>>>(mask, idx);
    convert_w_kernel<<<1024, 256, 0, stream>>>(Wq, Wk, Wqk);
    gather_convert_kernel<<<dim3(NSENT, BB), 256, 0, stream>>>(hs, idx, Wr, br, Xh, vbuf);
    gemm_qk_kernel<<<dim3(256 / 64, NSENT / 16, BB), 32, 0, stream>>>(Xh, Wqk, bq, bk, QK);
    finalize_kernel<<<BB, 128, 0, stream>>>(QK, vbuf, out_sent, out_seq);
}
